// GraphPPIS_14173392077177
// MI455X (gfx1250) — matrix operand order repaired and checked by reading, not compiled, not run on hardware
//
#include <hip/hip_runtime.h>
#include <hip/hip_bf16.h>
#include <cmath>

#define N_NODES 50000
#define N_EDGES 800000
#define NODE_F  54
#define HIDDEN  256
#define N_LAYERS 8

typedef __bf16 bf16x16 __attribute__((ext_vector_type(16)));
typedef float  f32x8   __attribute__((ext_vector_type(8)));
typedef float  f32x16  __attribute__((ext_vector_type(16)));
typedef unsigned int u32x4 __attribute__((ext_vector_type(4)));
typedef int          i32x4 __attribute__((ext_vector_type(4)));
typedef int          i32x8 __attribute__((ext_vector_type(8)));

__device__ __forceinline__ f32x8 wmma_bf16(bf16x16 a, bf16x16 b, f32x8 c) {
    // (neg_a, A, neg_b, B, c_mod, C, reuse_a, reuse_b)
    return __builtin_amdgcn_wmma_f32_16x16x32_bf16(false, a, false, b, (short)0, c, false, false);
}

// ---------------------------------------------------------------------------
// TDM: load a 2D fp32 tile [tile rows=128 x 32 elems] from global into LDS.
//  - data_size=4B, tensor_dim0=256 (row length), tensor_dim0_stride=256
//  - tensor_dim1 = rows remaining in tensor -> HW zero-fills OOB rows
//  - pad_enable: +4 dwords every 32 dwords  => LDS row stride 36 floats (144B)
// ---------------------------------------------------------------------------
#define LDSW 36              // LDS row stride in floats
#define TILE_BYTES (128 * LDSW * 4)

__device__ __forceinline__ void tdm_load_tile(unsigned lds_byte_addr,
                                              const float* gptr,
                                              unsigned rows_remaining) {
    unsigned long long ga = (unsigned long long)(uintptr_t)gptr;
    u32x4 g0;
    g0[0] = 1u;                                            // count=1, user desc
    g0[1] = lds_byte_addr;                                 // lds_addr
    g0[2] = (unsigned)(ga & 0xFFFFFFFFu);                  // global_addr[31:0]
    g0[3] = (unsigned)((ga >> 32) & 0x01FFFFFFu) | (2u << 30); // addr[56:32], type=2

    const unsigned td0 = 256u;          // tensor dim0 (elements)
    const unsigned td1 = rows_remaining;
    i32x8 g1;
    g1[0] = (int)((2u << 16)            // data_size = 4B
                | (1u << 20)            // pad_enable
                | (4u << 22)            // pad_interval: 32 dwords
                | (3u << 25));          // pad_amount: 4 dwords
    g1[1] = (int)((td0 & 0xFFFFu) << 16);                        // barrier_addr=0 | td0 lo16
    g1[2] = (int)(((td0 >> 16) & 0xFFFFu) | ((td1 & 0xFFFFu) << 16));
    g1[3] = (int)(((td1 >> 16) & 0xFFFFu) | (32u << 16));        // tile_dim0 = 32
    g1[4] = (int)128u;                                           // tile_dim1=128, tile_dim2=0
    g1[5] = (int)256u;                                           // tensor_dim0_stride lo32
    g1[6] = 0;                                                   // stride0 hi, stride1 lo
    g1[7] = 0;

    i32x4 z4 = {0, 0, 0, 0};
#if __has_include(<hip/amd_detail/amd_gfx1250_TDM.h>)
    i32x8 z8 = {0, 0, 0, 0, 0, 0, 0, 0};
    __builtin_amdgcn_tensor_load_to_lds(g0, g1, z4, z4, z8, 0);
#else
    __builtin_amdgcn_tensor_load_to_lds(g0, g1, z4, z4, 0);
#endif
}

// ---------------------------------------------------------------------------
// Pack convW (fp32 [L][512][256]) into WMMA B-fragment order, bf16.
// B 32x16 layout: lanes 0-15 -> K 0-7 & 16-23, lanes 16-31 -> K 8-15 & 24-31 (matches A).
// ---------------------------------------------------------------------------
__global__ __launch_bounds__(256) void packw_kernel(const float* __restrict__ convW,
                                                    __bf16* __restrict__ packW) {
    int id = blockIdx.x * 256 + threadIdx.x;   // 8*16*16*32*16 = 1048576 total
    int j    = id & 15;
    int lane = (id >> 4) & 31;
    int nt   = (id >> 9) & 15;
    int kb   = (id >> 13) & 15;
    int l    = id >> 17;
    int half = lane >> 4;
    int n = nt * 16 + (lane & 15);
    int k = kb * 32 + (j < 8 ? half * 8 + j : 16 + half * 8 + (j - 8));
    packW[id] = (__bf16)convW[((size_t)l * 512 + k) * 256 + n];
}

// i0 = relu(x @ W0 + b0)   [50000 x 54] @ [54 x 256]
__global__ __launch_bounds__(256) void init_embed_kernel(const float* __restrict__ x,
                                                         const float* __restrict__ W0,
                                                         const float* __restrict__ b0,
                                                         float* __restrict__ i0) {
    __shared__ float xs[NODE_F];
    int node = blockIdx.x;
    int j = threadIdx.x;
    if (j < NODE_F) xs[j] = x[node * NODE_F + j];
    __syncthreads();
    float s = b0[j];
#pragma unroll 6
    for (int k = 0; k < NODE_F; ++k) s += xs[k] * W0[k * HIDDEN + j];
    i0[(size_t)node * HIDDEN + j] = s > 0.f ? s : 0.f;
}

__global__ __launch_bounds__(256) void zero_kernel(float4* __restrict__ p, int n4) {
    int i = blockIdx.x * blockDim.x + threadIdx.x;
    if (i < n4) p[i] = make_float4(0.f, 0.f, 0.f, 0.f);
}

// agg[dst] += h[src] * w   — one wave per edge, 8 floats per lane
__global__ __launch_bounds__(256) void scatter_kernel(const float* __restrict__ h,
                                                      const float* __restrict__ ew,
                                                      const int* __restrict__ src,
                                                      const int* __restrict__ dst,
                                                      float* __restrict__ agg) {
    int e = blockIdx.x * 8 + (threadIdx.x >> 5);
    if (e >= N_EDGES) return;
    int lane = threadIdx.x & 31;
    int s = src[e], d = dst[e];
    float w = ew[e];
    const float4* hr = (const float4*)&h[(size_t)s * HIDDEN];
    float* ar = &agg[(size_t)d * HIDDEN];
#pragma unroll
    for (int i = 0; i < 2; ++i) {
        float4 v = hr[lane * 2 + i];
        int b = (lane * 2 + i) * 4;
        unsafeAtomicAdd(&ar[b + 0], v.x * w);
        unsafeAtomicAdd(&ar[b + 1], v.y * w);
        unsafeAtomicAdd(&ar[b + 2], v.z * w);
        unsafeAtomicAdd(&ar[b + 3], v.w * w);
    }
}

// ---------------------------------------------------------------------------
// Fused layer GEMM:  h = relu(theta*(concat(agg,i0)@Wl) + (1-theta)*(0.3*agg+0.7*i0) + i0)
// Block tile 128x128, 8 waves; wave = 2x4 grid of 16x16 WMMA tiles; K=512 in 16 steps of 32.
// A tile staged fp32 by TDM (double-buffered, overlapped with WMMA), converted to bf16
// in-register at fragment load. B pre-packed bf16 fragments (L2-resident).
// ---------------------------------------------------------------------------
__global__ __launch_bounds__(256) void gcn_gemm_kernel(const float* __restrict__ agg,
                                                       const float* __restrict__ i0,
                                                       const __bf16* __restrict__ packW,
                                                       float* __restrict__ h,
                                                       float theta) {
    __shared__ __align__(16) float a_lds[2][128 * LDSW];

    const int t    = threadIdx.x;
    const int lane = t & 31;
    const int w    = t >> 5;
    const int wm   = w >> 1;           // 0..3 -> M offset wm*32
    const int wn   = w & 1;            // 0..1 -> N offset wn*64
    const int m0   = blockIdx.x * 128;
    const int half = lane >> 4;
    const int r16  = lane & 15;
    const unsigned rows_rem = (unsigned)(N_NODES - m0);
    const unsigned lds0 = (unsigned)(unsigned long long)&a_lds[0][0];

    f32x8 acc[2][4];
#pragma unroll
    for (int mi = 0; mi < 2; ++mi)
#pragma unroll
        for (int ni = 0; ni < 4; ++ni)
            acc[mi][ni] = (f32x8){0.f, 0.f, 0.f, 0.f, 0.f, 0.f, 0.f, 0.f};

    // kb in [0,8): agg columns kb*32 ; kb in [8,16): i0 columns (kb-8)*32
    if (w == 0)
        tdm_load_tile(lds0, agg + (size_t)m0 * HIDDEN, rows_rem);

    for (int kb = 0; kb < 16; ++kb) {
        const int buf = kb & 1;
        if (w == 0) __builtin_amdgcn_s_wait_tensorcnt(0);   // tile kb resident
        __syncthreads();                                    // visible to all; prior reads done
        if (w == 0 && kb < 15) {                            // prefetch tile kb+1 into other buffer
            int kn = kb + 1;
            const float* srcp = (kn < 8) ? (agg + (size_t)m0 * HIDDEN + kn * 32)
                                         : (i0 + (size_t)m0 * HIDDEN + (kn - 8) * 32);
            tdm_load_tile(lds0 + (unsigned)((buf ^ 1) * TILE_BYTES), srcp, rows_rem);
        }

        // A fragments: 16-bit A 16x32 layout — lanes 0-15: K 0-7 & 16-23; lanes 16-31: K 8-15 & 24-31
        bf16x16 afrag[2];
#pragma unroll
        for (int mi = 0; mi < 2; ++mi) {
            int row = wm * 32 + mi * 16 + r16;
            const float* rp = &a_lds[buf][row * LDSW];
            union { float4 q[4]; f32x16 f; } af;
            af.q[0] = *(const float4*)(rp + half * 8);
            af.q[1] = *(const float4*)(rp + half * 8 + 4);
            af.q[2] = *(const float4*)(rp + 16 + half * 8);
            af.q[3] = *(const float4*)(rp + 20 + half * 8);
            afrag[mi] = __builtin_convertvector(af.f, bf16x16);   // v_cvt_pk_bf16_f32
        }
        // B fragments: pre-packed, 32 contiguous bytes per lane
        bf16x16 bfrag[4];
#pragma unroll
        for (int ni = 0; ni < 4; ++ni) {
            int nt = blockIdx.y * 8 + wn * 4 + ni;
            const uint4* p = (const uint4*)&packW[(((size_t)kb * 16 + nt) * 32 + lane) * 16];
            union { uint4 u[2]; bf16x16 v; } cv;
            cv.u[0] = p[0]; cv.u[1] = p[1];
            bfrag[ni] = cv.v;
        }
#pragma unroll
        for (int mi = 0; mi < 2; ++mi)
#pragma unroll
            for (int ni = 0; ni < 4; ++ni)
                acc[mi][ni] = wmma_bf16(afrag[mi], bfrag[ni], acc[mi][ni]);
    }

    // Epilogue: C/D layout — VGPR i: lanes 0-15 M=i, lanes 16-31 M=8+i; N = lane&15
    const float omt = 1.f - theta;
#pragma unroll
    for (int mi = 0; mi < 2; ++mi) {
#pragma unroll
        for (int i = 0; i < 8; ++i) {
            int m = m0 + wm * 32 + mi * 16 + half * 8 + i;
            if (m >= N_NODES) continue;
#pragma unroll
            for (int ni = 0; ni < 4; ++ni) {
                int n = blockIdx.y * 128 + wn * 64 + ni * 16 + r16;
                size_t idx = (size_t)m * HIDDEN + n;
                float a = agg[idx], e = i0[idx];
                float r = 0.3f * a + 0.7f * e;
                float v = theta * acc[mi][ni][i] + omt * r + e;
                h[idx] = v > 0.f ? v : 0.f;
            }
        }
    }
}

// out = h @ W1 + b1   [50000 x 256] @ [256 x 2] — one wave per node
__global__ __launch_bounds__(256) void out_kernel(const float* __restrict__ h,
                                                  const float* __restrict__ W1,
                                                  const float* __restrict__ b1,
                                                  float* __restrict__ out) {
    int node = blockIdx.x * 8 + (threadIdx.x >> 5);
    if (node >= N_NODES) return;
    int lane = threadIdx.x & 31;
    float s0 = 0.f, s1 = 0.f;
#pragma unroll
    for (int i = 0; i < 8; ++i) {
        int k = lane + i * 32;
        float hv = h[(size_t)node * HIDDEN + k];
        s0 += hv * W1[k * 2 + 0];
        s1 += hv * W1[k * 2 + 1];
    }
#pragma unroll
    for (int off = 16; off > 0; off >>= 1) {
        s0 += __shfl_down(s0, off, 32);
        s1 += __shfl_down(s1, off, 32);
    }
    if (lane == 0) {
        out[node * 2 + 0] = s0 + b1[0];
        out[node * 2 + 1] = s1 + b1[1];
    }
}

extern "C" void kernel_launch(void* const* d_in, const int* in_sizes, int n_in,
                              void* d_out, int out_size, void* d_ws, size_t ws_size,
                              hipStream_t stream) {
    const float* x      = (const float*)d_in[0];
    const float* edge_w = (const float*)d_in[1];
    const float* W0     = (const float*)d_in[2];
    const float* b0     = (const float*)d_in[3];
    const float* convW  = (const float*)d_in[4];
    const float* W1     = (const float*)d_in[5];
    const float* b1     = (const float*)d_in[6];
    const int*   src    = (const int*)d_in[7];
    const int*   dst    = (const int*)d_in[8];
    float*       out    = (float*)d_out;

    const size_t nodeBytes = (size_t)N_NODES * HIDDEN * sizeof(float);  // 51.2 MB
    char* ws = (char*)d_ws;
    float* i0  = (float*)(ws);
    float* h   = (float*)(ws + nodeBytes);
    float* agg = (float*)(ws + 2 * nodeBytes);
    __bf16* packW = (__bf16*)(ws + 3 * nodeBytes);                      // 2 MB bf16

    // one-time weight pack + initial embedding
    packw_kernel<<<4096, 256, 0, stream>>>(convW, packW);
    init_embed_kernel<<<N_NODES, 256, 0, stream>>>(x, W0, b0, i0);

    const int n4 = N_NODES * HIDDEN / 4;
    dim3 gemmGrid((N_NODES + 127) / 128, HIDDEN / 128);

    const float* hcur = i0;
    for (int l = 0; l < N_LAYERS; ++l) {
        zero_kernel<<<(n4 + 255) / 256, 256, 0, stream>>>((float4*)agg, n4);
        scatter_kernel<<<N_EDGES / 8, 256, 0, stream>>>(hcur, edge_w, src, dst, agg);
        float theta = fminf(1.0f, logf(1.5f / (float)(l + 1) + 1.0f));
        gcn_gemm_kernel<<<gemmGrid, 256, 0, stream>>>(agg, i0, packW + (size_t)l * 16 * 16 * 32 * 16, h, theta);
        hcur = h;
    }
    out_kernel<<<(N_NODES + 7) / 8, 256, 0, stream>>>(h, W1, b1, out);
}
